// DecoderModel_63883343561184
// MI455X (gfx1250) — matrix operand order repaired and checked by reading, not compiled, not run on hardware
//
#include <hip/hip_runtime.h>
#include <hip/hip_bf16.h>

typedef unsigned short u16;
typedef __bf16 bf16_t;
typedef bf16_t v16bf __attribute__((ext_vector_type(16)));
typedef bf16_t v8bf  __attribute__((ext_vector_type(8)));
typedef float  v8f   __attribute__((ext_vector_type(8)));

#define KB 128
#define KP 196
#define KT 20
#define KIT 10
#define KENC 2048
#define KH 512
#define KE 512
#define KA 512
#define KVOCAB 7464
#define KTRI 5283

__device__ __forceinline__ u16 f2bf(float f) {
  union { float f; unsigned u; } v; v.f = f;
  unsigned u = v.u;
  return (u16)((u + 0x7FFFu + ((u >> 16) & 1u)) >> 16);   // RNE
}
__device__ __forceinline__ float bf2f(u16 h) {
  union { unsigned u; float f; } v; v.u = ((unsigned)h) << 16; return v.f;
}
__device__ __forceinline__ float sigm(float x) { return 1.f / (1.f + __expf(-x)); }

// ---------------------------------------------------------------------------
// NT bf16 GEMM: C[M,N] (+)= A[M,K](bf16) * W[N,K](bf16)^T + bias, * rowscale
// wave tile 16Mx64N (4x v_wmma_f32_16x16x32_bf16).
// Block = 8 waves stacked along M -> block tile 128M x 64N, so each weight
// tile is streamed from L2/HBM exactly once per GEMM (weights dominate bytes).
// K-step is split into a load phase (all 10 b128 loads in flight) and a
// compute phase (4 WMMAs) so s_wait_loadcnt can retire loads progressively.
// ---------------------------------------------------------------------------
__global__ __launch_bounds__(256) void gemm_nt_bf16(
    const u16* __restrict__ Au, int lda,
    const u16* __restrict__ Wu, int ldw,
    const float* __restrict__ bias,
    float* __restrict__ C, long ldc,
    const float* __restrict__ rowscale,
    int M, int N, int K, int acc)
{
  const int wid  = threadIdx.x >> 5;
  const int lane = threadIdx.x & 31;
  const int kh   = lane >> 4;          // which half-lane group
  const int l16  = lane & 15;
  const int m0   = (blockIdx.y << 7) + (wid << 4);
  const int nw   = blockIdx.x << 6;
  if (m0 >= M) return;

  const bf16_t* A = (const bf16_t*)Au;
  const bf16_t* W = (const bf16_t*)Wu;

  v8f zero = {0.f,0.f,0.f,0.f,0.f,0.f,0.f,0.f};
  v8f cacc[4];
#pragma unroll
  for (int j = 0; j < 4; ++j) cacc[j] = zero;

  if (acc) {
#pragma unroll
    for (int j = 0; j < 4; ++j) {
      int n = nw + j * 16 + l16;
      if (n < N) {
#pragma unroll
        for (int r = 0; r < 8; ++r)
          cacc[j][r] = C[(long)(m0 + kh * 8 + r) * ldc + n];
      }
    }
  }

  const bf16_t* Arow = A + (long)(m0 + l16) * lda;
  const bf16_t* Wp[4];
#pragma unroll
  for (int j = 0; j < 4; ++j) {
    int n  = nw + j * 16 + l16;
    int nc = (n < N) ? n : (N - 1);          // clamp: junk lanes never stored
    Wp[j] = W + (long)nc * ldw + kh * 16;
  }

  for (int k0 = 0; k0 < K; k0 += 32) {
    // ---- load phase: issue all fragment loads before any consumption ----
    union { v16bf v; v8bf h[2]; } au;
    au.h[0] = *(const v8bf*)(Arow + k0 + kh * 16);       // K = k0+kh*16  .. +7
    au.h[1] = *(const v8bf*)(Arow + k0 + kh * 16 + 8);   // K = k0+kh*16+8.. +15
    v16bf bfrag[4];
#pragma unroll
    for (int j = 0; j < 4; ++j) {
      bfrag[j] = *(const v16bf*)(Wp[j] + k0);            // K = k0+kh*16 .. +15
      __builtin_prefetch(Wp[j] + k0 + 128, 0, 1);        // 4 k-steps (256B) ahead
    }
    // ---- compute phase ----
#pragma unroll
    for (int j = 0; j < 4; ++j) {
      cacc[j] = __builtin_amdgcn_wmma_f32_16x16x32_bf16(
          false, au.v, false, bfrag[j], (short)0, cacc[j], false, false);
    }
  }

#pragma unroll
  for (int j = 0; j < 4; ++j) {
    int n = nw + j * 16 + l16;
    if (n >= N) continue;
    float bv = bias ? bias[n] : 0.f;
#pragma unroll
    for (int r = 0; r < 8; ++r) {
      int row = m0 + kh * 8 + r;
      float val = cacc[j][r] + bv;
      if (rowscale) val *= rowscale[row];
      C[(long)row * ldc + n] = val;
    }
  }
}

// ---------------------------------------------------------------------------
// elementwise f32 -> bf16
__global__ void k_cvt(const float* __restrict__ src, u16* __restrict__ dst, long n) {
  long i = (long)blockIdx.x * 256 + threadIdx.x;
  if (i < n) dst[i] = f2bf(src[i]);
}

// maskT[t*B+b] = (t < lengths[b])
__global__ void k_masks(const int* __restrict__ len, float* __restrict__ maskT) {
  int i = blockIdx.x * 256 + threadIdx.x;
  if (i < KT * KB) {
    int t = i / KB, b = i % KB;
    maskT[i] = (t < len[b]) ? 1.f : 0.f;
  }
}

// mean over P of encoder_out -> bf16 (B,ENC)
__global__ void k_mean_enc(const float* __restrict__ enc, u16* __restrict__ out) {
  int b = blockIdx.x;
  int e = blockIdx.y * 256 + threadIdx.x;
  const float* base = enc + (long)b * KP * KENC + e;
  float s = 0.f;
  for (int p = 0; p < KP; ++p) s += base[(long)p * KENC];
  out[b * KENC + e] = f2bf(s * (1.f / KP));
}

// mean over IT of rwe -> bf16 (B,E)
__global__ void k_mean_rwe(const float* __restrict__ rwe, u16* __restrict__ out) {
  int b = blockIdx.x;
  int e = blockIdx.y * 256 + threadIdx.x;
  const float* base = rwe + (long)b * KIT * KE + e;
  float s = 0.f;
  for (int i = 0; i < KIT; ++i) s += base[i * KE];
  out[b * KE + e] = f2bf(s * (1.f / KIT));
}

// att[b,p] = sum_a relu(enc2[b,p,a]+e1[b,a])*attfW[a] + attfb  (one wave per (b,p))
__global__ __launch_bounds__(256) void k_att_scores(
    const float* __restrict__ enc2, const float* __restrict__ e1,
    const float* __restrict__ attfW, const float* __restrict__ attfb,
    float* __restrict__ att)
{
  int idx  = blockIdx.x * 8 + (threadIdx.x >> 5);
  int lane = threadIdx.x & 31;
  if (idx >= KB * KP) return;
  int b = idx / KP;
  const float* row = enc2 + (long)idx * KA;
  const float* e1r = e1 + b * KA;
  float s = 0.f;
  for (int a = lane; a < KA; a += 32) {
    float v = row[a] + e1r[a];
    v = v > 0.f ? v : 0.f;
    s += v * attfW[a];
  }
  for (int off = 16; off > 0; off >>= 1) s += __shfl_xor(s, off, 32);
  if (lane == 0) att[idx] = s + attfb[0];
}

// softmax over P; writes alpha (scratch) and masked alpha into d_out
__global__ __launch_bounds__(256) void k_softmax_alpha(
    const float* __restrict__ att, float* __restrict__ alpha,
    float* __restrict__ out_alphas, const float* __restrict__ mrow, int t)
{
  int b = blockIdx.x, tid = threadIdx.x;
  __shared__ float sv[KP];
  __shared__ float red[256];
  float v = -1e30f;
  if (tid < KP) { sv[tid] = att[b * KP + tid]; v = sv[tid]; }
  red[tid] = v; __syncthreads();
  for (int s = 128; s > 0; s >>= 1) { if (tid < s) red[tid] = fmaxf(red[tid], red[tid + s]); __syncthreads(); }
  float mx = red[0]; __syncthreads();
  float e = 0.f;
  if (tid < KP) { e = __expf(sv[tid] - mx); sv[tid] = e; }
  red[tid] = e; __syncthreads();
  for (int s = 128; s > 0; s >>= 1) { if (tid < s) red[tid] += red[tid + s]; __syncthreads(); }
  float inv = 1.f / red[0];
  if (tid < KP) {
    float al = sv[tid] * inv;
    alpha[b * KP + tid] = al;
    out_alphas[(long)b * KT * KP + (long)t * KP + tid] = al * mrow[b];
  }
}

// awe[b,e] = sigmoid(gpre[b,e]) * sum_p alpha[b,p]*enc_bf[b,p,e] -> x_cap[:,1024+e] bf16
__global__ __launch_bounds__(256) void k_awe(
    const float* __restrict__ alpha, const u16* __restrict__ encbf,
    const float* __restrict__ gpre, u16* __restrict__ xcap)
{
  int b = blockIdx.x, tid = threadIdx.x;
  __shared__ float al[KP];
  if (tid < KP) al[tid] = alpha[b * KP + tid];
  __syncthreads();
  float acc[8] = {0.f,0.f,0.f,0.f,0.f,0.f,0.f,0.f};
  const u16* base = encbf + (long)b * KP * KENC;
  for (int p = 0; p < KP; ++p) {
    float ap = al[p];
    const u16* r = base + (long)p * KENC;
#pragma unroll
    for (int j = 0; j < 8; ++j) acc[j] += ap * bf2f(r[tid + j * 256]);
  }
#pragma unroll
  for (int j = 0; j < 8; ++j) {
    int e = tid + j * 256;
    float g = sigm(gpre[b * KENC + e]);
    xcap[(long)b * 3072 + 1024 + e] = f2bf(acc[j] * g);
  }
}

// gsc = softmax(leaky_relu(rwe . h1)); rws = rwe^T gsc -> x_ds[:,512:1024] bf16
__global__ __launch_bounds__(256) void k_gsc_rws(
    const float* __restrict__ rwe, const float* __restrict__ h1,
    u16* __restrict__ xds)
{
  int b = blockIdx.x, tid = threadIdx.x;
  __shared__ float h1s[KE];
  __shared__ float red[256];
  __shared__ float sc[KIT];
  h1s[tid] = h1[b * KH + tid];
  h1s[tid + 256] = h1[b * KH + tid + 256];
  __syncthreads();
  const float* rb = rwe + (long)b * KIT * KE;
  for (int i = 0; i < KIT; ++i) {
    float s = rb[i * KE + tid] * h1s[tid] + rb[i * KE + tid + 256] * h1s[tid + 256];
    red[tid] = s; __syncthreads();
    for (int st = 128; st > 0; st >>= 1) { if (tid < st) red[tid] += red[tid + st]; __syncthreads(); }
    if (tid == 0) { float v = red[0]; sc[i] = v > 0.f ? v : 0.2f * v; }
    __syncthreads();
  }
  float mx = -1e30f;
  for (int i = 0; i < KIT; ++i) mx = fmaxf(mx, sc[i]);
  float w[KIT]; float sum = 0.f;
  for (int i = 0; i < KIT; ++i) { w[i] = __expf(sc[i] - mx); sum += w[i]; }
  float inv = 1.f / sum;
  for (int e = tid; e < KE; e += 256) {
    float a = 0.f;
    for (int i = 0; i < KIT; ++i) a += rb[i * KE + e] * w[i];
    xds[(long)b * 1024 + 512 + e] = f2bf(a * inv);
  }
}

// x_ds[:,0:512] = tri_emb_bf16[idx[b,t]]
__global__ void k_gather_temb(const int* __restrict__ idx, const u16* __restrict__ tri,
                              u16* __restrict__ xds, int t) {
  int b = blockIdx.x;
  int id = idx[b * KT + t];
  for (int e = threadIdx.x; e < KE; e += 256)
    xds[(long)b * 1024 + e] = tri[(long)id * KE + e];
}

// x_cap[:,0:512] = bf16(captions_embedding[b,t,:])
__global__ void k_cvt_cap(const float* __restrict__ cap, u16* __restrict__ xcap, int t) {
  int b = blockIdx.x;
  int e = blockIdx.y * 256 + threadIdx.x;
  xcap[(long)b * 3072 + e] = f2bf(cap[((long)b * KT + t) * KE + e]);
}

// LSTM pointwise: gates (B,4H) -> masked carry update + bf16 copies of hn
__global__ __launch_bounds__(256) void k_lstm(
    const float* __restrict__ g, float* __restrict__ h, float* __restrict__ c,
    u16* __restrict__ h_bf, u16* __restrict__ hn_bf0, int ld0,
    u16* __restrict__ hn_bf1, int ld1, const float* __restrict__ mrow)
{
  int b = blockIdx.x;
  int j = blockIdx.y * 256 + threadIdx.x;   // < 512
  const float* gb = g + (long)b * (4 * KH);
  float iv = sigm(gb[j]);
  float fv = sigm(gb[KH + j]);
  float gv = tanhf(gb[2 * KH + j]);
  float ov = sigm(gb[3 * KH + j]);
  float cold = c[b * KH + j];
  float cn = fv * cold + iv * gv;
  float hn = ov * tanhf(cn);
  bool m = mrow[b] > 0.5f;
  float hold = h[b * KH + j];
  float hnew = m ? hn : hold;
  float cnew = m ? cn : cold;
  h[b * KH + j] = hnew;
  c[b * KH + j] = cnew;
  h_bf[b * KH + j] = f2bf(hnew);
  hn_bf0[(long)b * ld0 + j] = f2bf(hn);
  if (hn_bf1) hn_bf1[(long)b * ld1 + j] = f2bf(hn);
}

// end[b,t] = (exp(vp_last) + exp(tp_last)) * mask
__global__ __launch_bounds__(256) void k_end(
    const u16* __restrict__ h1n, const u16* __restrict__ h2n,
    const float* __restrict__ fctW, const float* __restrict__ fctb,
    const float* __restrict__ fcvW, const float* __restrict__ fcvb,
    const float* __restrict__ mrow, float* __restrict__ out_end, int t)
{
  int b = blockIdx.x, tid = threadIdx.x;
  __shared__ float red[256];
  const float* wt = fctW + (long)(KTRI - 1) * KH;
  const float* wv = fcvW + (long)(KVOCAB - 1) * KH;
  float s1 = bf2f(h1n[b * KH + tid]) * wt[tid] + bf2f(h1n[b * KH + 256 + tid]) * wt[256 + tid];
  float s2 = bf2f(h2n[b * KH + tid]) * wv[tid] + bf2f(h2n[b * KH + 256 + tid]) * wv[256 + tid];
  red[tid] = s1; __syncthreads();
  for (int s = 128; s > 0; s >>= 1) { if (tid < s) red[tid] += red[tid + s]; __syncthreads(); }
  float d1 = red[0]; __syncthreads();
  red[tid] = s2; __syncthreads();
  for (int s = 128; s > 0; s >>= 1) { if (tid < s) red[tid] += red[tid + s]; __syncthreads(); }
  if (tid == 0) {
    float d2 = red[0];
    out_end[b * KT + t] = (__expf(d2 + fcvb[KVOCAB - 1]) + __expf(d1 + fctb[KTRI - 1])) * mrow[b];
  }
}

// ---------------------------------------------------------------------------
extern "C" void kernel_launch(void* const* d_in, const int* in_sizes, int n_in,
                              void* d_out, int out_size, void* d_ws, size_t ws_size,
                              hipStream_t stream) {
  const float* enc    = (const float*)d_in[0];
  const int*   triIdx = (const int*)d_in[1];
  const int*   lens   = (const int*)d_in[2];
  const float* capEmb = (const float*)d_in[3];
  const float* rwe    = (const float*)d_in[4];
  const float* att1W = (const float*)d_in[5];  const float* att1b = (const float*)d_in[6];
  const float* att2W = (const float*)d_in[7];  const float* att2b = (const float*)d_in[8];
  const float* attfW = (const float*)d_in[9];  const float* attfb = (const float*)d_in[10];
  const float* betaW = (const float*)d_in[11]; const float* betab = (const float*)d_in[12];
  const float* ih1W  = (const float*)d_in[13]; const float* ih1b  = (const float*)d_in[14];
  const float* ic1W  = (const float*)d_in[15]; const float* ic1b  = (const float*)d_in[16];
  const float* ih2W  = (const float*)d_in[17]; const float* ih2b  = (const float*)d_in[18];
  const float* ic2W  = (const float*)d_in[19]; const float* ic2b  = (const float*)d_in[20];
  const float* fctW  = (const float*)d_in[21]; const float* fctb  = (const float*)d_in[22];
  const float* fcvW  = (const float*)d_in[23]; const float* fcvb  = (const float*)d_in[24];
  const float* dsWih = (const float*)d_in[25];
  const float* dsWhh = (const float*)d_in[26];
  const float* dsbih = (const float*)d_in[27];
  const float* dsbhh = (const float*)d_in[28];
  const float* capWih = (const float*)d_in[29];
  const float* capWhh = (const float*)d_in[30];
  const float* capbih = (const float*)d_in[31];
  const float* capbhh = (const float*)d_in[32];
  const float* triEmb = (const float*)d_in[33];

  // output slices (return order: cap_preds, ds_preds, end_score, alphas)
  float* out_cap    = (float*)d_out;
  float* out_ds     = out_cap + (long)KB * KT * KVOCAB;
  float* out_end    = out_ds + (long)KB * KT * KTRI;
  float* out_alphas = out_end + (long)KB * KT;

  // workspace allocator
  char* wp = (char*)d_ws;
  auto alloc = [&](size_t bytes) -> void* {
    void* r = (void*)wp;
    wp += (bytes + 255) & ~(size_t)255;
    return r;
  };
  u16* enc_bf   = (u16*)alloc((size_t)KB * KP * KENC * 2);
  u16* wAtt1    = (u16*)alloc((size_t)KA * KH * 2);
  u16* wAtt2    = (u16*)alloc((size_t)KA * KENC * 2);
  u16* wBeta    = (u16*)alloc((size_t)KENC * KH * 2);
  u16* wIh1     = (u16*)alloc((size_t)KH * KENC * 2);
  u16* wIc1     = (u16*)alloc((size_t)KH * KENC * 2);
  u16* wIh2     = (u16*)alloc((size_t)KH * KE * 2);
  u16* wIc2     = (u16*)alloc((size_t)KH * KE * 2);
  u16* wFct     = (u16*)alloc((size_t)KTRI * KH * 2);
  u16* wFcv     = (u16*)alloc((size_t)KVOCAB * KH * 2);
  u16* wDsWih   = (u16*)alloc((size_t)4 * KH * 1024 * 2);
  u16* wDsWhh   = (u16*)alloc((size_t)4 * KH * KH * 2);
  u16* wCapWih  = (u16*)alloc((size_t)4 * KH * 3072 * 2);
  u16* wCapWhh  = (u16*)alloc((size_t)4 * KH * KH * 2);
  u16* wTri     = (u16*)alloc((size_t)KTRI * KE * 2);
  u16* meanEncB = (u16*)alloc((size_t)KB * KENC * 2);
  u16* meanRweB = (u16*)alloc((size_t)KB * KE * 2);
  u16* h1_bf    = (u16*)alloc((size_t)KB * KH * 2);
  u16* h2_bf    = (u16*)alloc((size_t)KB * KH * 2);
  u16* h1n_bf   = (u16*)alloc((size_t)KB * KH * 2);
  u16* h2n_bf   = (u16*)alloc((size_t)KB * KH * 2);
  u16* x_ds     = (u16*)alloc((size_t)KB * 1024 * 2);
  u16* x_cap    = (u16*)alloc((size_t)KB * 3072 * 2);
  float* enc2   = (float*)alloc((size_t)KB * KP * KA * 4);
  float* e1     = (float*)alloc((size_t)KB * KA * 4);
  float* att    = (float*)alloc((size_t)KB * KP * 4);
  float* alpha  = (float*)alloc((size_t)KB * KP * 4);
  float* gpre   = (float*)alloc((size_t)KB * KENC * 4);
  float* g_ds   = (float*)alloc((size_t)KB * 4 * KH * 4);
  float* g_cap  = (float*)alloc((size_t)KB * 4 * KH * 4);
  float* h1f    = (float*)alloc((size_t)KB * KH * 4);
  float* c1f    = (float*)alloc((size_t)KB * KH * 4);
  float* h2f    = (float*)alloc((size_t)KB * KH * 4);
  float* c2f    = (float*)alloc((size_t)KB * KH * 4);
  float* maskT  = (float*)alloc((size_t)KT * KB * 4);

  auto cvt = [&](const float* s, u16* d, long n) {
    k_cvt<<<(unsigned)((n + 255) / 256), 256, 0, stream>>>(s, d, n);
  };
  auto gemm = [&](const u16* A, int lda, const u16* W, int ldw, const float* bias,
                  float* C, long ldc, const float* rs, int M, int N, int K, int acc) {
    dim3 g((N + 63) / 64, (M + 127) / 128);
    gemm_nt_bf16<<<g, 256, 0, stream>>>(A, lda, W, ldw, bias, C, ldc, rs, M, N, K, acc);
  };

  // ---- one-time conversions + setup ----
  cvt(enc,    enc_bf,  (long)KB * KP * KENC);
  cvt(att1W,  wAtt1,   (long)KA * KH);
  cvt(att2W,  wAtt2,   (long)KA * KENC);
  cvt(betaW,  wBeta,   (long)KENC * KH);
  cvt(ih1W,   wIh1,    (long)KH * KENC);
  cvt(ic1W,   wIc1,    (long)KH * KENC);
  cvt(ih2W,   wIh2,    (long)KH * KE);
  cvt(ic2W,   wIc2,    (long)KH * KE);
  cvt(fctW,   wFct,    (long)KTRI * KH);
  cvt(fcvW,   wFcv,    (long)KVOCAB * KH);
  cvt(dsWih,  wDsWih,  (long)4 * KH * 1024);
  cvt(dsWhh,  wDsWhh,  (long)4 * KH * KH);
  cvt(capWih, wCapWih, (long)4 * KH * 3072);
  cvt(capWhh, wCapWhh, (long)4 * KH * KH);
  cvt(triEmb, wTri,    (long)KTRI * KE);
  k_masks<<<(KT * KB + 255) / 256, 256, 0, stream>>>(lens, maskT);
  k_mean_enc<<<dim3(KB, KENC / 256), 256, 0, stream>>>(enc, meanEncB);
  k_mean_rwe<<<dim3(KB, KE / 256), 256, 0, stream>>>(rwe, meanRweB);

  // initial states
  gemm(meanEncB, KENC, wIh1, KENC, ih1b, h2f, KH, nullptr, KB, KH, KENC, 0);
  gemm(meanEncB, KENC, wIc1, KENC, ic1b, c2f, KH, nullptr, KB, KH, KENC, 0);
  gemm(meanRweB, KE,   wIh2, KE,   ih2b, h1f, KH, nullptr, KB, KH, KE, 0);
  gemm(meanRweB, KE,   wIc2, KE,   ic2b, c1f, KH, nullptr, KB, KH, KE, 0);
  cvt(h1f, h1_bf, (long)KB * KH);
  cvt(h2f, h2_bf, (long)KB * KH);

  // enc2 = encoder_out @ att2.W^T + b   (M = B*P = 196*128)
  gemm(enc_bf, KENC, wAtt2, KENC, att2b, enc2, KA, nullptr, KB * KP, KA, KENC, 0);

  // ---- time loop ----
  for (int t = 0; t < KT; ++t) {
    const float* mrow = maskT + t * KB;

    gemm(h2_bf, KH, wAtt1, KH, att1b, e1, KA, nullptr, KB, KA, KH, 0);
    k_att_scores<<<(KB * KP + 7) / 8, 256, 0, stream>>>(enc2, e1, attfW, attfb, att);
    k_softmax_alpha<<<KB, 256, 0, stream>>>(att, alpha, out_alphas, mrow, t);
    gemm(h2_bf, KH, wBeta, KH, betab, gpre, KENC, nullptr, KB, KENC, KH, 0);
    k_gsc_rws<<<KB, 256, 0, stream>>>(rwe, h1f, x_ds);
    k_gather_temb<<<KB, 256, 0, stream>>>(triIdx, wTri, x_ds, t);
    k_awe<<<KB, 256, 0, stream>>>(alpha, enc_bf, gpre, x_cap);

    // ds LSTM gates
    gemm(x_ds, 1024, wDsWih, 1024, dsbih, g_ds, 4 * KH, nullptr, KB, 4 * KH, 1024, 0);
    gemm(h1_bf, KH,  wDsWhh, KH,   dsbhh, g_ds, 4 * KH, nullptr, KB, 4 * KH, KH, 1);
    k_lstm<<<dim3(KB, 2), 256, 0, stream>>>(g_ds, h1f, c1f, h1_bf,
                                            x_cap + 512, 3072, h1n_bf, KH, mrow);

    k_cvt_cap<<<dim3(KB, 2), 256, 0, stream>>>(capEmb, x_cap, t);

    // cap LSTM gates
    gemm(x_cap, 3072, wCapWih, 3072, capbih, g_cap, 4 * KH, nullptr, KB, 4 * KH, 3072, 0);
    gemm(h2_bf, KH,   wCapWhh, KH,   capbhh, g_cap, 4 * KH, nullptr, KB, 4 * KH, KH, 1);
    k_lstm<<<dim3(KB, 2), 256, 0, stream>>>(g_cap, h2f, c2f, h2_bf,
                                            h2n_bf, KH, nullptr, 0, mrow);

    // output heads, masked, written straight into d_out (strided)
    gemm(h1n_bf, KH, wFct, KH, fctb, out_ds + (long)t * KTRI,
         (long)KT * KTRI, mrow, KB, KTRI, KH, 0);
    gemm(h2n_bf, KH, wFcv, KH, fcvb, out_cap + (long)t * KVOCAB,
         (long)KT * KVOCAB, mrow, KB, KVOCAB, KH, 0);
    k_end<<<KB, 256, 0, stream>>>(h1n_bf, h2n_bf, fctW, fctb, fcvW, fcvb,
                                  mrow, out_end, t);
  }
  (void)in_sizes; (void)n_in; (void)out_size; (void)ws_size;
}
